// Block_4793183502627
// MI455X (gfx1250) — matrix operand order repaired and checked by reading, not compiled, not run on hardware
//
#include <hip/hip_runtime.h>
#include <math.h>

// MI455X / gfx1250: wave32, WMMA 16x16x32 bf16 -> f32 accumulate.
// Fused transformer block. Compute-bound (~880 GFLOP vs ~330MB HBM), so all
// large GEMMs go through v_wmma_f32_16x16x32_bf16. Attention (N=17) is 0.3%
// of FLOPs -> plain fp32 VALU. GEMM tile staging uses the gfx1250 async
// global->LDS DMA path (ASYNCcnt) when the builtins are available.

typedef __attribute__((ext_vector_type(16))) __bf16 v16bf;
typedef __attribute__((ext_vector_type(8)))  __bf16 v8bf;
typedef __attribute__((ext_vector_type(8)))  float  v8f;
typedef __attribute__((ext_vector_type(4)))  int    v4i;

#define DIMC   1024
#define NTOK   17
#define NHEADS 16
#define HEADD  64
#define HIDC   4096
#define MROWS  (2048 * NTOK)   // 34816 = 272 * 128

// ---- gfx1250 async global->LDS (GLOBAL_LOAD_ASYNC_TO_LDS_B128, ASYNCcnt) ----
#if defined(__has_builtin)
#  if __has_builtin(__builtin_amdgcn_global_load_async_to_lds_b128)
#    define HAVE_ASYNC_LDS 1
#  else
#    define HAVE_ASYNC_LDS 0
#  endif
#else
#  define HAVE_ASYNC_LDS 0
#endif

#if HAVE_ASYNC_LDS
__device__ __forceinline__ void async_cp16(const void* g, void* l) {
  // builtin signature: (v4i addrspace(1)*, v4i addrspace(3)*, imm offset, imm cpol)
  __builtin_amdgcn_global_load_async_to_lds_b128(
      (__attribute__((address_space(1))) v4i*)g,
      (__attribute__((address_space(3))) v4i*)l, 0, 0);
}
#if __has_builtin(__builtin_amdgcn_s_wait_asynccnt)
#define WAIT_ASYNC() __builtin_amdgcn_s_wait_asynccnt(0)
#else
#define WAIT_ASYNC() asm volatile("s_wait_asynccnt 0x0" ::: "memory")
#endif
#endif

// ---------------------------------------------------------------------------
// Weight convert + transpose: in fp32 [K,N] row-major -> out bf16 [N,K]
// ---------------------------------------------------------------------------
__global__ void convert_transpose_kernel(const float* __restrict__ in,
                                         __bf16* __restrict__ out,
                                         int K, int N) {
  long total = (long)K * N;
  for (long idx = (long)blockIdx.x * blockDim.x + threadIdx.x; idx < total;
       idx += (long)gridDim.x * blockDim.x) {
    long k = idx / N;
    long n = idx - k * N;
    out[n * K + k] = (__bf16)in[idx];
  }
}

// ---------------------------------------------------------------------------
// Edge bias: bias[j] = b_edge[j] + sum_i e[i] * W_edge[i*289 + j]   (289 = 17*17)
// ---------------------------------------------------------------------------
__global__ void edge_bias_kernel(const float* __restrict__ e,
                                 const float* __restrict__ W,
                                 const float* __restrict__ b,
                                 float* __restrict__ out) {
  int j = blockIdx.x * blockDim.x + threadIdx.x;
  if (j >= NTOK * NTOK) return;
  float s = b[j];
  for (int i = 0; i < NTOK * NTOK; ++i) s += e[i] * W[i * (NTOK * NTOK) + j];
  out[j] = s;
}

// ---------------------------------------------------------------------------
// LayerNorm over last dim (1024), output bf16. One block (128 thr) per row.
// ---------------------------------------------------------------------------
__global__ __launch_bounds__(128)
void ln_bf16_kernel(const float* __restrict__ x, const float* __restrict__ g,
                    const float* __restrict__ b, __bf16* __restrict__ out) {
  const long row = blockIdx.x;
  const float* xr = x + row * DIMC;
  __shared__ float red[128];
  __shared__ float mu_s, rs_s;

  float v[8];
  float s = 0.f, s2 = 0.f;
#pragma unroll
  for (int i = 0; i < 8; ++i) {
    v[i] = xr[threadIdx.x + i * 128];
    s += v[i];
    s2 += v[i] * v[i];
  }
  red[threadIdx.x] = s;
  __syncthreads();
  for (int off = 64; off > 0; off >>= 1) {
    if (threadIdx.x < off) red[threadIdx.x] += red[threadIdx.x + off];
    __syncthreads();
  }
  if (threadIdx.x == 0) mu_s = red[0] * (1.0f / DIMC);
  __syncthreads();
  const float mu = mu_s;
  __syncthreads();
  red[threadIdx.x] = s2;
  __syncthreads();
  for (int off = 64; off > 0; off >>= 1) {
    if (threadIdx.x < off) red[threadIdx.x] += red[threadIdx.x + off];
    __syncthreads();
  }
  if (threadIdx.x == 0) {
    float var = red[0] * (1.0f / DIMC) - mu * mu;
    rs_s = rsqrtf(var + 1e-5f);
  }
  __syncthreads();
  const float rs = rs_s;
#pragma unroll
  for (int i = 0; i < 8; ++i) {
    int c = threadIdx.x + i * 128;
    out[row * DIMC + c] = (__bf16)(((v[i] - mu) * rs) * g[c] + b[c]);
  }
}

// ---------------------------------------------------------------------------
// bf16 WMMA GEMM:  C[M,Nc] = epilogue(A[M,K] @ Bt[Nc,K]^T + bias (+resid))
//   EPI 0: store bf16, no bias            (qkv)
//   EPI 1: +bias, exact GELU, store bf16  (fc1)
//   EPI 2: +bias, +resid, store fp32      (proj, fc2)
// Block: 256 threads (8 waves), 128x128 tile, K-chunk 32, double-buffered LDS.
// Tiles staged via async global->LDS DMA (overlapped with WMMA) when available.
// ---------------------------------------------------------------------------
template <int EPI>
__global__ __launch_bounds__(256)
void gemm_bf16_kernel(const __bf16* __restrict__ A, const __bf16* __restrict__ Bt,
                      const float* __restrict__ bias, const float* __restrict__ resid,
                      float* __restrict__ Cf, __bf16* __restrict__ Cb,
                      int M, int Nc, int K) {
  __shared__ __align__(16) __bf16 As[2][128 * 32];
  __shared__ __align__(16) __bf16 Bs[2][128 * 32];

  const int tid  = threadIdx.x;
  const int wave = tid >> 5;
  const int lane = tid & 31;
  const int wm = wave >> 1;      // 0..3
  const int wn = wave & 1;       // 0..1
  const int h  = lane >> 4;      // lane half
  const int l16 = lane & 15;

  const long rowA0 = (long)blockIdx.y * 128;
  const long rowB0 = (long)blockIdx.x * 128;

  // load mapping: 512 chunks of 16B per tile; chunk c -> row c>>2, kpos (c&3)*8
  const int cA0 = tid, cA1 = tid + 256;
  const int rA0 = cA0 >> 2, kA0 = (cA0 & 3) * 8;
  const int rA1 = cA1 >> 2, kA1 = (cA1 & 3) * 8;

  v8f acc[2][4];
#pragma unroll
  for (int mi = 0; mi < 2; ++mi)
#pragma unroll
    for (int ni = 0; ni < 4; ++ni) acc[mi][ni] = (v8f){0.f,0.f,0.f,0.f,0.f,0.f,0.f,0.f};

  const int nk = K >> 5;

#if HAVE_ASYNC_LDS
  // ---- async DMA double-buffer: tile k+1 in flight while tile k feeds WMMA ----
  async_cp16(A  + (rowA0 + rA0) * K + kA0, &As[0][rA0 * 32 + kA0]);
  async_cp16(A  + (rowA0 + rA1) * K + kA1, &As[0][rA1 * 32 + kA1]);
  async_cp16(Bt + (rowB0 + rA0) * K + kA0, &Bs[0][rA0 * 32 + kA0]);
  async_cp16(Bt + (rowB0 + rA1) * K + kA1, &Bs[0][rA1 * 32 + kA1]);
  WAIT_ASYNC();
  __syncthreads();
#else
  float4 la0, la1, lb0, lb1;
  la0 = *(const float4*)(A  + (rowA0 + rA0) * K + kA0);
  la1 = *(const float4*)(A  + (rowA0 + rA1) * K + kA1);
  lb0 = *(const float4*)(Bt + (rowB0 + rA0) * K + kA0);
  lb1 = *(const float4*)(Bt + (rowB0 + rA1) * K + kA1);
  *(float4*)&As[0][rA0 * 32 + kA0] = la0;
  *(float4*)&As[0][rA1 * 32 + kA1] = la1;
  *(float4*)&Bs[0][rA0 * 32 + kA0] = lb0;
  *(float4*)&Bs[0][rA1 * 32 + kA1] = lb1;
  __syncthreads();
#endif

  for (int kt = 0; kt < nk; ++kt) {
    const int cur = kt & 1;
    const bool hasNext = (kt + 1) < nk;

    if (hasNext) {
      const long k0 = (long)(kt + 1) << 5;
      const int nxt = cur ^ 1;
#if HAVE_ASYNC_LDS
      async_cp16(A  + (rowA0 + rA0) * K + k0 + kA0, &As[nxt][rA0 * 32 + kA0]);
      async_cp16(A  + (rowA0 + rA1) * K + k0 + kA1, &As[nxt][rA1 * 32 + kA1]);
      async_cp16(Bt + (rowB0 + rA0) * K + k0 + kA0, &Bs[nxt][rA0 * 32 + kA0]);
      async_cp16(Bt + (rowB0 + rA1) * K + k0 + kA1, &Bs[nxt][rA1 * 32 + kA1]);
#else
      la0 = *(const float4*)(A  + (rowA0 + rA0) * K + k0 + kA0);
      la1 = *(const float4*)(A  + (rowA0 + rA1) * K + k0 + kA1);
      lb0 = *(const float4*)(Bt + (rowB0 + rA0) * K + k0 + kA0);
      lb1 = *(const float4*)(Bt + (rowB0 + rA1) * K + k0 + kA1);
      if (kt + 2 < nk) {  // warm L2/WGP$ for K+2 tile -> global_prefetch_b8
        const long k2 = (long)(kt + 2) << 5;
        __builtin_prefetch(A  + (rowA0 + rA0) * K + k2 + kA0, 0, 1);
        __builtin_prefetch(Bt + (rowB0 + rA0) * K + k2 + kA0, 0, 1);
      }
      (void)nxt;
#endif
    }

#pragma unroll
    for (int mi = 0; mi < 2; ++mi) {
      const int m = wm * 32 + mi * 16 + l16;
      const v8bf a0 = *(const v8bf*)&As[cur][m * 32 + h * 8];
      const v8bf a1 = *(const v8bf*)&As[cur][m * 32 + 16 + h * 8];
      const v16bf a = __builtin_shufflevector(a0, a1, 0,1,2,3,4,5,6,7,8,9,10,11,12,13,14,15);
#pragma unroll
      for (int ni = 0; ni < 4; ++ni) {
        const int n = wn * 64 + ni * 16 + l16;
        const v8bf b0 = *(const v8bf*)&Bs[cur][n * 32 + h * 8];
        const v8bf b1 = *(const v8bf*)&Bs[cur][n * 32 + 16 + h * 8];
        const v16bf b = __builtin_shufflevector(b0, b1, 0,1,2,3,4,5,6,7,8,9,10,11,12,13,14,15);
        acc[mi][ni] = __builtin_amdgcn_wmma_f32_16x16x32_bf16(
            false, a, false, b, (short)0, acc[mi][ni], false, false);
      }
    }

    if (hasNext) {
#if HAVE_ASYNC_LDS
      WAIT_ASYNC();
#else
      const int nxt = cur ^ 1;
      *(float4*)&As[nxt][rA0 * 32 + kA0] = la0;
      *(float4*)&As[nxt][rA1 * 32 + kA1] = la1;
      *(float4*)&Bs[nxt][rA0 * 32 + kA0] = lb0;
      *(float4*)&Bs[nxt][rA1 * 32 + kA1] = lb1;
#endif
    }
    __syncthreads();
  }

  // epilogue: C/D layout (ISA 7.12.2): VGPR r, lane half h -> row h*8+r, col lane&15
  const long crow0 = rowA0 + wm * 32;
#pragma unroll
  for (int mi = 0; mi < 2; ++mi) {
#pragma unroll
    for (int ni = 0; ni < 4; ++ni) {
      const int c = (int)rowB0 + wn * 64 + ni * 16 + l16;
      const float bv = (EPI == 0) ? 0.f : bias[c];
#pragma unroll
      for (int r = 0; r < 8; ++r) {
        const long row = crow0 + mi * 16 + h * 8 + r;
        float val = acc[mi][ni][r] + bv;
        if (EPI == 1) val = 0.5f * val * (1.f + erff(val * 0.70710678118654752f));
        if (EPI == 2) {
          val += resid[row * Nc + c];
          Cf[row * Nc + c] = val;
        } else {
          Cb[row * Nc + c] = (__bf16)val;
        }
      }
    }
  }
}

// ---------------------------------------------------------------------------
// Attention per (b, head). N=17, D=64. fp32 VALU (0.3% of total FLOPs).
// qkv rows: [b*17+i][3072]: q at h*64, k at 1024+h*64, v at 2048+h*64.
// ---------------------------------------------------------------------------
__global__ __launch_bounds__(320)
void attn_kernel(const __bf16* __restrict__ qkv, const float* __restrict__ bias2d,
                 __bf16* __restrict__ o) {
  const int bh = blockIdx.x;
  const int b  = bh >> 4;
  const int hd = bh & 15;
  __shared__ float q[NTOK][HEADD], k[NTOK][HEADD], v[NTOK][HEADD];
  __shared__ float p[NTOK][NTOK];

  const __bf16* base = qkv + (long)b * NTOK * 3072 + hd * HEADD;
  for (int idx = threadIdx.x; idx < NTOK * HEADD; idx += 320) {
    const int i = idx >> 6, d = idx & 63;
    q[i][d] = (float)base[(long)i * 3072 + d];
    k[i][d] = (float)base[(long)i * 3072 + 1024 + d];
    v[i][d] = (float)base[(long)i * 3072 + 2048 + d];
  }
  __syncthreads();

  if (threadIdx.x < NTOK * NTOK) {
    const int i = threadIdx.x / NTOK, j = threadIdx.x % NTOK;
    float s = 0.f;
#pragma unroll
    for (int d = 0; d < HEADD; ++d) s += q[i][d] * k[j][d];
    p[i][j] = s * 0.125f + bias2d[threadIdx.x];  // SCALE = 64^-0.5
  }
  __syncthreads();

  if (threadIdx.x < NTOK) {
    const int i = threadIdx.x;
    float m = -1e30f;
    for (int j = 0; j < NTOK; ++j) m = fmaxf(m, p[i][j]);
    float sum = 0.f;
    for (int j = 0; j < NTOK; ++j) { float e = __expf(p[i][j] - m); p[i][j] = e; sum += e; }
    const float inv = 1.f / sum;
    for (int j = 0; j < NTOK; ++j) p[i][j] *= inv;
  }
  __syncthreads();

  for (int idx = threadIdx.x; idx < NTOK * HEADD; idx += 320) {
    const int i = idx >> 6, d = idx & 63;
    float s = 0.f;
#pragma unroll
    for (int j = 0; j < NTOK; ++j) s += p[i][j] * v[j][d];
    o[((long)b * NTOK + i) * DIMC + hd * HEADD + d] = (__bf16)s;
  }
}

// ---------------------------------------------------------------------------
// Launch
// ---------------------------------------------------------------------------
extern "C" void kernel_launch(void* const* d_in, const int* in_sizes, int n_in,
                              void* d_out, int out_size, void* d_ws, size_t ws_size,
                              hipStream_t stream) {
  const float* x      = (const float*)d_in[0];
  const float* edge   = (const float*)d_in[1];
  const float* W_qkv  = (const float*)d_in[2];
  const float* W_proj = (const float*)d_in[3];
  const float* b_proj = (const float*)d_in[4];
  const float* W_edge = (const float*)d_in[5];
  const float* b_edge = (const float*)d_in[6];
  const float* g1     = (const float*)d_in[7];
  const float* b1     = (const float*)d_in[8];
  const float* g2     = (const float*)d_in[9];
  const float* b2     = (const float*)d_in[10];
  const float* W_fc1  = (const float*)d_in[11];
  const float* b_fc1  = (const float*)d_in[12];
  const float* W_fc2  = (const float*)d_in[13];
  const float* b_fc2  = (const float*)d_in[14];
  float* out = (float*)d_out;

  char* wsp = (char*)d_ws;
  auto wsalloc = [&](size_t bytes) -> char* {
    char* p = wsp;
    wsp += (bytes + 255) & ~(size_t)255;
    return p;
  };
  float*  bias2d = (float*)wsalloc((size_t)NTOK * NTOK * 4);
  __bf16* WqkvT  = (__bf16*)wsalloc((size_t)DIMC * 3072 * 2);
  __bf16* WprojT = (__bf16*)wsalloc((size_t)DIMC * DIMC * 2);
  __bf16* Wfc1T  = (__bf16*)wsalloc((size_t)DIMC * HIDC * 2);
  __bf16* Wfc2T  = (__bf16*)wsalloc((size_t)HIDC * DIMC * 2);
  __bf16* xn     = (__bf16*)wsalloc((size_t)MROWS * DIMC * 2);
  __bf16* qkvb   = (__bf16*)wsalloc((size_t)MROWS * 3072 * 2);
  __bf16* ob     = (__bf16*)wsalloc((size_t)MROWS * DIMC * 2);
  float*  x1     = (float*)wsalloc((size_t)MROWS * DIMC * 4);
  __bf16* x1n    = (__bf16*)wsalloc((size_t)MROWS * DIMC * 2);
  __bf16* h2     = (__bf16*)wsalloc((size_t)MROWS * HIDC * 2);

  // 1. weights -> bf16 transposed [N,K]
  convert_transpose_kernel<<<2048, 256, 0, stream>>>(W_qkv,  WqkvT,  DIMC, 3072);
  convert_transpose_kernel<<<1024, 256, 0, stream>>>(W_proj, WprojT, DIMC, DIMC);
  convert_transpose_kernel<<<2048, 256, 0, stream>>>(W_fc1,  Wfc1T,  DIMC, HIDC);
  convert_transpose_kernel<<<2048, 256, 0, stream>>>(W_fc2,  Wfc2T,  HIDC, DIMC);

  // 2. edge bias (17x17)
  edge_bias_kernel<<<1, 320, 0, stream>>>(edge, W_edge, b_edge, bias2d);

  // 3. LN1 -> bf16
  ln_bf16_kernel<<<MROWS, 128, 0, stream>>>(x, g1, b1, xn);

  // 4. qkv = xn @ W_qkv           [34816, 3072]
  {
    dim3 grid(3072 / 128, MROWS / 128);
    gemm_bf16_kernel<0><<<grid, 256, 0, stream>>>(xn, WqkvT, nullptr, nullptr,
                                                  nullptr, qkvb, MROWS, 3072, DIMC);
  }

  // 5. attention with edge bias
  attn_kernel<<<2048 * NHEADS, 320, 0, stream>>>(qkvb, bias2d, ob);

  // 6. x1 = x + o @ W_proj + b_proj   (fp32)
  {
    dim3 grid(DIMC / 128, MROWS / 128);
    gemm_bf16_kernel<2><<<grid, 256, 0, stream>>>(ob, WprojT, b_proj, x,
                                                  x1, nullptr, MROWS, DIMC, DIMC);
  }

  // 7. LN2 -> bf16
  ln_bf16_kernel<<<MROWS, 128, 0, stream>>>(x1, g2, b2, x1n);

  // 8. h2 = gelu(x1n @ W_fc1 + b_fc1)  [34816, 4096] bf16
  {
    dim3 grid(HIDC / 128, MROWS / 128);
    gemm_bf16_kernel<1><<<grid, 256, 0, stream>>>(x1n, Wfc1T, b_fc1, nullptr,
                                                  nullptr, h2, MROWS, HIDC, DIMC);
  }

  // 9. out = x1 + h2 @ W_fc2 + b_fc2   (fp32)
  {
    dim3 grid(DIMC / 128, MROWS / 128);
    gemm_bf16_kernel<2><<<grid, 256, 0, stream>>>(h2, Wfc2T, b_fc2, x1,
                                                  out, nullptr, MROWS, DIMC, HIDC);
  }
}
